// PytorchMeanshiftFusion_77103252898445
// MI455X (gfx1250) — compile-verified, matrix operand order checked by reading, not run on hardware
//
#include <hip/hip_runtime.h>
#include <hip/hip_bf16.h>

// Sizes fixed by the reference
#define N_PTS  6144
#define D_DIM  64
#define FF_DIM 256
#define EPSF   1e-5f

typedef _Float16 h16;
typedef __attribute__((ext_vector_type(16))) _Float16 v16h;
typedef __attribute__((ext_vector_type(8)))  _Float16 v8h_t;
typedef __attribute__((ext_vector_type(8)))  float    v8f;

// ---------------------------------------------------------------------------
// Fragment loader: CDNA5 16-bit A/B fragment from a row-major f16 row with
// contiguous K. Per lane: halfs[0..7] = row[k0+ko .. +7], halfs[8..15] =
// row[k0+ko+16 .. +7] where ko = 8 for lanes 16..31. Two b128 loads.
// ---------------------------------------------------------------------------
__device__ __forceinline__ v16h load_frag(const h16* __restrict__ row, int k0, int ko) {
  const v8h_t lo = *(const v8h_t*)(row + k0 + ko);
  const v8h_t hv = *(const v8h_t*)(row + k0 + ko + 16);
  v16h r;
#pragma unroll
  for (int i = 0; i < 8; ++i) { r[i] = lo[i]; r[8 + i] = hv[i]; }
  return r;
}

__device__ __forceinline__ v8f wmma16(v16h a, v16h b, v8f c) {
  return __builtin_amdgcn_wmma_f32_16x16x32_f16(false, a, false, b, (short)0, c,
                                                false, false);
}

// ---------------------------------------------------------------------------
// Setup: f32 -> f16 convert, and f32 [R][C] -> f16 [C][R] transpose
// ---------------------------------------------------------------------------
__global__ void cvt_f32_h16_kernel(const float* __restrict__ src,
                                   h16* __restrict__ dst, int n) {
  int i = blockIdx.x * blockDim.x + threadIdx.x;
  if (i < n) dst[i] = (h16)src[i];
}

__global__ void transpose_f32_h16_kernel(const float* __restrict__ src,
                                         h16* __restrict__ dst, int R, int C) {
  int i = blockIdx.x * blockDim.x + threadIdx.x;
  if (i >= R * C) return;
  int r = i / C, c = i - r * C;
  dst[c * R + r] = (h16)src[i];
}

// ---------------------------------------------------------------------------
// qk = rg_fea @ qp_w + qp_b   -> f16 [N][64]
// ---------------------------------------------------------------------------
__global__ __launch_bounds__(128) void qk_kernel(const h16* __restrict__ rgh,
                                                 const h16* __restrict__ wT,
                                                 const float* __restrict__ bias,
                                                 h16* __restrict__ qkh) {
  const int wave = (blockIdx.x * blockDim.x + threadIdx.x) >> 5;
  const int lane = threadIdx.x & 31;
  const int q0 = wave * 16;
  const int n = lane & 15;
  const int ko = (lane >> 4) << 3;
  const h16* arow = rgh + (q0 + n) * D_DIM;
  v16h a0 = load_frag(arow, 0, ko);
  v16h a1 = load_frag(arow, 32, ko);
#pragma unroll
  for (int t = 0; t < 4; ++t) {
    const h16* brow = wT + (t * 16 + n) * D_DIM;
    v8f acc = {};
    acc = wmma16(a0, load_frag(brow, 0, ko), acc);
    acc = wmma16(a1, load_frag(brow, 32, ko), acc);
    const int col = t * 16 + n;
    const float bv = bias[col];
#pragma unroll
    for (int v = 0; v < 8; ++v)
      qkh[(q0 + v + ko) * D_DIM + col] = (h16)(acc[v] + bv);
  }
}

// ---------------------------------------------------------------------------
// Fused flash attention + residual + LayerNorm (S^T / O^T formulation).
// Software-pipelined: next block's 8 fragments are issued right after the
// S-WMMAs so their latency overlaps softmax VALU + PV WMMAs.
// ---------------------------------------------------------------------------
__global__ __launch_bounds__(128) void flash_attn_ln_kernel(
    const h16* __restrict__ qkh,   // [N][64]
    const h16* __restrict__ vT,    // [64][N]  (x_fea transposed, f16)
    const float* __restrict__ x_fea,
    const float* __restrict__ ln_g, const float* __restrict__ ln_b,
    float* __restrict__ h1, h16* __restrict__ h1h) {
  const int wave = (blockIdx.x * blockDim.x + threadIdx.x) >> 5;
  const int lane = threadIdx.x & 31;
  const int q0 = wave * 16;
  const int n = lane & 15;
  const int ko = (lane >> 4) << 3;

  // Q^T B-fragments (column n = query q0+n), loaded once
  const h16* qrow = qkh + (q0 + n) * D_DIM;
  const v16h bq0 = load_frag(qrow, 0, ko);
  const v16h bq1 = load_frag(qrow, 32, ko);

  const h16* krow = qkh + n * D_DIM;  // + j0*64 per block
  const h16* vr0 = vT + (0 + n) * N_PTS;
  const h16* vr1 = vT + (16 + n) * N_PTS;
  const h16* vr2 = vT + (32 + n) * N_PTS;
  const h16* vr3 = vT + (48 + n) * N_PTS;

  v8f ot[4] = {};
  float mrow = -1e30f, lrow = 0.0f;

  // Pipeline prologue: fragments for block j0 = 0
  v16h k0a = load_frag(krow, 0, ko);
  v16h k0b = load_frag(krow, 32, ko);
  v16h k1a = load_frag(krow + 16 * D_DIM, 0, ko);
  v16h k1b = load_frag(krow + 16 * D_DIM, 32, ko);
  v16h va = load_frag(vr0, 0, ko);
  v16h vb = load_frag(vr1, 0, ko);
  v16h vc = load_frag(vr2, 0, ko);
  v16h vd = load_frag(vr3, 0, ko);

  for (int j0 = 0; j0 < N_PTS; j0 += 32) {
    const int jn = (j0 + 32 < N_PTS) ? (j0 + 32) : 0;  // last prefetch wraps (discarded)
    if (j0 + 128 < N_PTS)
      __builtin_prefetch(krow + (j0 + 128) * D_DIM, 0, 1);  // global_prefetch_b8

    // S^T tiles for current block
    v8f st0 = {}, st1 = {};
    st0 = wmma16(k0a, bq0, st0);
    st0 = wmma16(k0b, bq1, st0);
    st1 = wmma16(k1a, bq0, st1);
    st1 = wmma16(k1b, bq1, st1);

    // Issue next block's fragment loads (overlap with softmax + PV below)
    v16h nk0a = load_frag(krow + jn * D_DIM, 0, ko);
    v16h nk0b = load_frag(krow + jn * D_DIM, 32, ko);
    v16h nk1a = load_frag(krow + (jn + 16) * D_DIM, 0, ko);
    v16h nk1b = load_frag(krow + (jn + 16) * D_DIM, 32, ko);
    v16h nva = load_frag(vr0, jn, ko);
    v16h nvb = load_frag(vr1, jn, ko);
    v16h nvc = load_frag(vr2, jn, ko);
    v16h nvd = load_frag(vr3, jn, ko);

    // Online softmax (per-query = per-lane pair; one xor16 exchange)
    float mcur = -1e30f;
#pragma unroll
    for (int v = 0; v < 8; ++v) {
      st0[v] *= 0.125f;  // 1/sqrt(64)
      st1[v] *= 0.125f;
      mcur = fmaxf(mcur, fmaxf(st0[v], st1[v]));
    }
    mcur = fmaxf(mcur, __shfl_xor(mcur, 16, 32));
    const float mnew = fmaxf(mrow, mcur);
    const float alpha = __expf(mrow - mnew);
    float lsum = 0.0f;
    v16h pb;  // P^T B-fragment, direct from S^T C-layout (no shuffles)
#pragma unroll
    for (int v = 0; v < 8; ++v) {
      const float p0 = __expf(st0[v] - mnew);
      const float p1 = __expf(st1[v] - mnew);
      lsum += p0 + p1;
      pb[v] = (h16)p0;
      pb[8 + v] = (h16)p1;
    }
    lsum += __shfl_xor(lsum, 16, 32);
    lrow = lrow * alpha + lsum;
    mrow = mnew;
#pragma unroll
    for (int t = 0; t < 4; ++t)
#pragma unroll
      for (int v = 0; v < 8; ++v) ot[t][v] *= alpha;

    ot[0] = wmma16(va, pb, ot[0]);
    ot[1] = wmma16(vb, pb, ot[1]);
    ot[2] = wmma16(vc, pb, ot[2]);
    ot[3] = wmma16(vd, pb, ot[3]);

    // Rotate pipeline registers
    k0a = nk0a; k0b = nk0b; k1a = nk1a; k1b = nk1b;
    va = nva; vb = nvb; vc = nvc; vd = nvd;
  }

  // Normalize, residual, LayerNorm over D=64 (32 values in-lane + xor16)
  const float rinv = 1.0f / lrow;
  const float* xrow = x_fea + (q0 + n) * D_DIM;
  float r[4][8];
  float sum = 0.0f, sumsq = 0.0f;
#pragma unroll
  for (int t = 0; t < 4; ++t)
#pragma unroll
    for (int v = 0; v < 8; ++v) {
      const int d = t * 16 + ko + v;
      const float val = xrow[d] + ot[t][v] * rinv;
      r[t][v] = val;
      sum += val;
      sumsq += val * val;
    }
  sum += __shfl_xor(sum, 16, 32);
  sumsq += __shfl_xor(sumsq, 16, 32);
  const float mean = sum * (1.0f / 64.0f);
  const float var = sumsq * (1.0f / 64.0f) - mean * mean;
  const float inv = rsqrtf(var + EPSF);
  const int row = q0 + n;
#pragma unroll
  for (int t = 0; t < 4; ++t)
#pragma unroll
    for (int v = 0; v < 8; ++v) {
      const int d = t * 16 + ko + v;
      const float hvv = (r[t][v] - mean) * inv * ln_g[d] + ln_b[d];
      h1[row * D_DIM + d] = hvv;
      h1h[row * D_DIM + d] = (h16)hvv;
    }
}

// ---------------------------------------------------------------------------
// z1 = relu(h1 @ ff_w1 + b1)  -> f16 [N][256]
// ---------------------------------------------------------------------------
__global__ __launch_bounds__(128) void ff1_kernel(const h16* __restrict__ h1h,
                                                  const h16* __restrict__ w1T,
                                                  const float* __restrict__ b1,
                                                  h16* __restrict__ z1h) {
  const int wave = (blockIdx.x * blockDim.x + threadIdx.x) >> 5;
  const int lane = threadIdx.x & 31;
  const int q0 = wave * 16;
  const int n = lane & 15;
  const int ko = (lane >> 4) << 3;
  const h16* arow = h1h + (q0 + n) * D_DIM;
  v16h a0 = load_frag(arow, 0, ko);
  v16h a1 = load_frag(arow, 32, ko);
#pragma unroll
  for (int t = 0; t < 16; ++t) {
    const int col = t * 16 + n;
    const h16* brow = w1T + col * D_DIM;
    v8f acc = {};
    acc = wmma16(a0, load_frag(brow, 0, ko), acc);
    acc = wmma16(a1, load_frag(brow, 32, ko), acc);
    const float bv = b1[col];
#pragma unroll
    for (int v = 0; v < 8; ++v) {
      float z = fmaxf(acc[v] + bv, 0.0f);
      z1h[(q0 + v + ko) * FF_DIM + col] = (h16)z;
    }
  }
}

// ---------------------------------------------------------------------------
// h2 = LN(h1 + z1 @ ff_w2 + b2)
// ---------------------------------------------------------------------------
__global__ __launch_bounds__(128) void ff2_ln_kernel(
    const h16* __restrict__ z1h, const h16* __restrict__ w2T,
    const float* __restrict__ b2, const float* __restrict__ h1,
    const float* __restrict__ ln_g, const float* __restrict__ ln_b,
    float* __restrict__ h2, h16* __restrict__ h2h) {
  const int wave = (blockIdx.x * blockDim.x + threadIdx.x) >> 5;
  const int lane = threadIdx.x & 31;
  const int q0 = wave * 16;
  const int n = lane & 15;
  const int ko = (lane >> 4) << 3;
  const h16* arow = z1h + (q0 + n) * FF_DIM;
  v8f acc[4] = {};
#pragma unroll
  for (int kf = 0; kf < 8; ++kf) {
    v16h a = load_frag(arow, kf * 32, ko);
#pragma unroll
    for (int t = 0; t < 4; ++t) {
      const h16* brow = w2T + (t * 16 + n) * FF_DIM;
      acc[t] = wmma16(a, load_frag(brow, kf * 32, ko), acc[t]);
    }
  }
  float vals[4][8];
#pragma unroll
  for (int t = 0; t < 4; ++t) {
    const int col = t * 16 + n;
    const float bv = b2[col];
#pragma unroll
    for (int v = 0; v < 8; ++v) {
      const int row = q0 + v + ko;
      vals[t][v] = acc[t][v] + bv + h1[row * D_DIM + col];
    }
  }
#pragma unroll
  for (int v = 0; v < 8; ++v) {
    float s = vals[0][v] + vals[1][v] + vals[2][v] + vals[3][v];
    float ss = vals[0][v] * vals[0][v] + vals[1][v] * vals[1][v] +
               vals[2][v] * vals[2][v] + vals[3][v] * vals[3][v];
#pragma unroll
    for (int m = 1; m < 16; m <<= 1) {
      s += __shfl_xor(s, m, 32);
      ss += __shfl_xor(ss, m, 32);
    }
    const float mean = s * (1.0f / 64.0f);
    const float var = ss * (1.0f / 64.0f) - mean * mean;
    const float inv = rsqrtf(var + EPSF);
    const int row = q0 + v + ko;
#pragma unroll
    for (int t = 0; t < 4; ++t) {
      const int col = t * 16 + n;
      const float hvv = (vals[t][v] - mean) * inv * ln_g[col] + ln_b[col];
      h2[row * D_DIM + col] = hvv;
      h2h[row * D_DIM + col] = (h16)hvv;
    }
  }
}

// ---------------------------------------------------------------------------
// y = h2 @ bw_w1[it] + bw_b1[it]  -> f32 [N][64]
// ---------------------------------------------------------------------------
__global__ __launch_bounds__(128) void bw1_kernel(const h16* __restrict__ h2h,
                                                  const h16* __restrict__ wT,
                                                  const float* __restrict__ bias,
                                                  float* __restrict__ y) {
  const int wave = (blockIdx.x * blockDim.x + threadIdx.x) >> 5;
  const int lane = threadIdx.x & 31;
  const int q0 = wave * 16;
  const int n = lane & 15;
  const int ko = (lane >> 4) << 3;
  const h16* arow = h2h + (q0 + n) * D_DIM;
  v16h a0 = load_frag(arow, 0, ko);
  v16h a1 = load_frag(arow, 32, ko);
#pragma unroll
  for (int t = 0; t < 4; ++t) {
    const int col = t * 16 + n;
    const h16* brow = wT + col * D_DIM;
    v8f acc = {};
    acc = wmma16(a0, load_frag(brow, 0, ko), acc);
    acc = wmma16(a1, load_frag(brow, 32, ko), acc);
    const float bv = bias[col];
#pragma unroll
    for (int v = 0; v < 8; ++v)
      y[(q0 + v + ko) * D_DIM + col] = acc[v] + bv;
  }
}

// ---------------------------------------------------------------------------
// Deterministic BatchNorm column stats over N=6144 rows (two-pass, no atomics)
// ---------------------------------------------------------------------------
__global__ __launch_bounds__(256) void colstats_kernel(const float* __restrict__ y,
                                                       float* __restrict__ parts) {
  __shared__ float sb[4][64];
  __shared__ float ssb[4][64];
  const int col = threadIdx.x & 63;
  const int rs = threadIdx.x >> 6;
  const int r0 = blockIdx.x * 128;
  float s = 0.0f, ss = 0.0f;
  for (int r = r0 + rs; r < r0 + 128; r += 4) {
    const float v = y[r * 64 + col];
    s += v;
    ss += v * v;
  }
  sb[rs][col] = s;
  ssb[rs][col] = ss;
  __syncthreads();
  if (rs == 0) {
    parts[blockIdx.x * 64 + col] = sb[0][col] + sb[1][col] + sb[2][col] + sb[3][col];
    parts[48 * 64 + blockIdx.x * 64 + col] =
        ssb[0][col] + ssb[1][col] + ssb[2][col] + ssb[3][col];
  }
}

__global__ void bn_finalize_kernel(const float* __restrict__ parts,
                                   float* __restrict__ stats) {
  const int col = threadIdx.x;  // 64 threads
  float s = 0.0f, ss = 0.0f;
  for (int b = 0; b < 48; ++b) {
    s += parts[b * 64 + col];
    ss += parts[48 * 64 + b * 64 + col];
  }
  const float mean = s * (1.0f / (float)N_PTS);
  const float var = ss * (1.0f / (float)N_PTS) - mean * mean;
  stats[col] = mean;
  stats[64 + col] = rsqrtf(var + EPSF);
}

// ---------------------------------------------------------------------------
// z = relu(BN(y)); logits = z @ bw_w2 + bw_b2; w = softmax(logits)
// ---------------------------------------------------------------------------
__global__ __launch_bounds__(256) void bn_logits_kernel(
    const float* __restrict__ y, const float* __restrict__ stats,
    const float* __restrict__ g, const float* __restrict__ b,
    const float* __restrict__ w2, const float* __restrict__ b2,
    float* __restrict__ wout) {
  const int i = blockIdx.x * 256 + threadIdx.x;
  if (i >= N_PTS) return;
  float l0 = b2[0], l1 = b2[1], l2 = b2[2];
#pragma unroll 4
  for (int d = 0; d < 64; ++d) {
    float z = (y[i * 64 + d] - stats[d]) * stats[64 + d] * g[d] + b[d];
    z = fmaxf(z, 0.0f);
    l0 += z * w2[d * 3 + 0];
    l1 += z * w2[d * 3 + 1];
    l2 += z * w2[d * 3 + 2];
  }
  const float m = fmaxf(l0, fmaxf(l1, l2));
  const float e0 = __expf(l0 - m), e1 = __expf(l1 - m), e2 = __expf(l2 - m);
  const float inv = 1.0f / (e0 + e1 + e2);
  wout[i * 3 + 0] = e0 * inv;
  wout[i * 3 + 1] = e1 * inv;
  wout[i * 3 + 2] = e2 * inv;
}

// ---------------------------------------------------------------------------
// Mean-shift with CDNA5 async global->LDS staging of the point tile.
// The instruction offset applies to both global and LDS addresses (ISA §15.18.3
// GLOBAL_LOAD_ASYNC_TO_LDS pseudocode), so one LDS-offset VGPR + offset:4/8
// stages the 3 dwords of each point. ASYNCcnt drained before the barrier.
// ---------------------------------------------------------------------------
__global__ __launch_bounds__(256) void meanshift_kernel(
    const float* __restrict__ xin, const float* __restrict__ w,
    float* __restrict__ xout) {
  __shared__ float xs[256 * 3];
  const int j = blockIdx.x * 256 + threadIdx.x;
  const float xj0 = xin[j * 3 + 0];
  const float xj1 = xin[j * 3 + 1];
  const float xj2 = xin[j * 3 + 2];
  float dg0 = 0, dg1 = 0, dg2 = 0;
  float a0x = 0, a0y = 0, a0z = 0;
  float a1x = 0, a1y = 0, a1z = 0;
  float a2x = 0, a2y = 0, a2z = 0;
  const unsigned lds_off = (unsigned)(size_t)(&xs[threadIdx.x * 3]);
  for (int base = 0; base < N_PTS; base += 256) {
    const float* gp = xin + (size_t)(base + threadIdx.x) * 3;
    asm volatile(
        "global_load_async_to_lds_b32 %0, %1, off\n\t"
        "global_load_async_to_lds_b32 %0, %1, off offset:4\n\t"
        "global_load_async_to_lds_b32 %0, %1, off offset:8\n\t"
        "s_wait_asynccnt 0x0"
        :
        : "v"(lds_off), "v"(gp)
        : "memory");
    __syncthreads();
    for (int t = 0; t < 256; ++t) {
      const float px = xs[t * 3 + 0];
      const float py = xs[t * 3 + 1];
      const float pz = xs[t * 3 + 2];
      const float dx = px - xj0, dy = py - xj1, dz = pz - xj2;
      const float d2 = dx * dx + dy * dy + dz * dz;
      if (d2 <= 1.0f) {
        if (d2 <= 0.04f) { dg0 += 1.0f; a0x += px; a0y += py; a0z += pz; }
        else if (d2 <= 0.25f) { dg1 += 1.0f; a1x += px; a1y += py; a1z += pz; }
        else { dg2 += 1.0f; a2x += px; a2y += py; a2z += pz; }
      }
    }
    __syncthreads();
  }
  // cumulative rings -> per-bandwidth sums
  const float D0 = dg0, D1 = dg0 + dg1, D2 = dg0 + dg1 + dg2;
  const float S0x = a0x, S1x = a0x + a1x, S2x = a0x + a1x + a2x;
  const float S0y = a0y, S1y = a0y + a1y, S2y = a0y + a1y + a2y;
  const float S0z = a0z, S1z = a0z + a1z, S2z = a0z + a1z + a2z;
  const float w0 = w[j * 3 + 0], w1 = w[j * 3 + 1], w2 = w[j * 3 + 2];
  const float winv = 1.0f / (w0 + w1 + w2);
  const float c0 = w0 / D0, c1 = w1 / D1, c2 = w2 / D2;
  xout[j * 3 + 0] = (S0x * c0 + S1x * c1 + S2x * c2) * winv;
  xout[j * 3 + 1] = (S0y * c0 + S1y * c1 + S2y * c2) * winv;
  xout[j * 3 + 2] = (S0z * c0 + S1z * c1 + S2z * c2) * winv;
}

// ---------------------------------------------------------------------------
// Host driver
// ---------------------------------------------------------------------------
extern "C" void kernel_launch(void* const* d_in, const int* in_sizes, int n_in,
                              void* d_out, int out_size, void* d_ws, size_t ws_size,
                              hipStream_t stream) {
  (void)in_sizes; (void)n_in; (void)out_size; (void)ws_size;
  const float* x      = (const float*)d_in[0];
  const float* rg_fea = (const float*)d_in[1];
  const float* x_fea  = (const float*)d_in[2];
  const float* qp_w   = (const float*)d_in[3];
  const float* qp_b   = (const float*)d_in[4];
  const float* ln_g   = (const float*)d_in[5];
  const float* ln_b   = (const float*)d_in[6];
  const float* ff_w1  = (const float*)d_in[7];
  const float* ff_b1  = (const float*)d_in[8];
  const float* ff_w2  = (const float*)d_in[9];
  const float* ff_b2  = (const float*)d_in[10];
  const float* bw_w1  = (const float*)d_in[11];
  const float* bw_b1  = (const float*)d_in[12];
  const float* bn_g   = (const float*)d_in[13];
  const float* bn_b   = (const float*)d_in[14];
  const float* bw_w2  = (const float*)d_in[15];
  const float* bw_b2  = (const float*)d_in[16];

  // Workspace carve (256B aligned)
  uintptr_t p = (uintptr_t)d_ws;
  auto carve = [&](size_t bytes) -> void* {
    void* r = (void*)p;
    p += (bytes + 255) & ~(size_t)255;
    return r;
  };
  h16*   rgh   = (h16*)carve((size_t)N_PTS * D_DIM * 2);
  h16*   qkh   = (h16*)carve((size_t)N_PTS * D_DIM * 2);
  h16*   vT    = (h16*)carve((size_t)D_DIM * N_PTS * 2);
  h16*   qpwT  = (h16*)carve((size_t)D_DIM * D_DIM * 2);
  h16*   w1T   = (h16*)carve((size_t)FF_DIM * D_DIM * 2);
  h16*   w2T   = (h16*)carve((size_t)D_DIM * FF_DIM * 2);
  h16*   bwT   = (h16*)carve((size_t)2 * D_DIM * D_DIM * 2);
  float* h1    = (float*)carve((size_t)N_PTS * D_DIM * 4);
  h16*   h1h   = (h16*)carve((size_t)N_PTS * D_DIM * 2);
  h16*   z1h   = (h16*)carve((size_t)N_PTS * FF_DIM * 2);
  float* h2    = (float*)carve((size_t)N_PTS * D_DIM * 4);
  h16*   h2h   = (h16*)carve((size_t)N_PTS * D_DIM * 2);
  float* ybuf  = (float*)carve((size_t)N_PTS * D_DIM * 4);
  float* parts = (float*)carve((size_t)48 * 64 * 2 * 4);
  float* stats = (float*)carve((size_t)2 * 64 * 4);
  float* wprob = (float*)carve((size_t)N_PTS * 3 * 4);
  float* xtmp  = (float*)carve((size_t)N_PTS * 3 * 4);

  const int WMMA_BLOCKS = (N_PTS / 16) / 4;  // 384 waves, 4 waves/block = 96

  // Setup conversions
  cvt_f32_h16_kernel<<<(N_PTS * D_DIM + 255) / 256, 256, 0, stream>>>(rg_fea, rgh, N_PTS * D_DIM);
  transpose_f32_h16_kernel<<<(N_PTS * D_DIM + 255) / 256, 256, 0, stream>>>(x_fea, vT, N_PTS, D_DIM);
  transpose_f32_h16_kernel<<<(D_DIM * D_DIM + 255) / 256, 256, 0, stream>>>(qp_w, qpwT, D_DIM, D_DIM);
  transpose_f32_h16_kernel<<<(D_DIM * FF_DIM + 255) / 256, 256, 0, stream>>>(ff_w1, w1T, D_DIM, FF_DIM);
  transpose_f32_h16_kernel<<<(FF_DIM * D_DIM + 255) / 256, 256, 0, stream>>>(ff_w2, w2T, FF_DIM, D_DIM);
  transpose_f32_h16_kernel<<<(D_DIM * D_DIM + 255) / 256, 256, 0, stream>>>(bw_w1, bwT, D_DIM, D_DIM);
  transpose_f32_h16_kernel<<<(D_DIM * D_DIM + 255) / 256, 256, 0, stream>>>(bw_w1 + 4096, bwT + 4096, D_DIM, D_DIM);

  // Attention + FFN
  qk_kernel<<<WMMA_BLOCKS, 128, 0, stream>>>(rgh, qpwT, qp_b, qkh);
  flash_attn_ln_kernel<<<WMMA_BLOCKS, 128, 0, stream>>>(qkh, vT, x_fea, ln_g, ln_b, h1, h1h);
  ff1_kernel<<<WMMA_BLOCKS, 128, 0, stream>>>(h1h, w1T, ff_b1, z1h);
  ff2_ln_kernel<<<WMMA_BLOCKS, 128, 0, stream>>>(z1h, w2T, ff_b2, h1, ln_g, ln_b, h2, h2h);

  // Mean-shift iterations
  for (int it = 0; it < 2; ++it) {
    bw1_kernel<<<WMMA_BLOCKS, 128, 0, stream>>>(h2h, bwT + it * 4096, bw_b1 + it * 64, ybuf);
    colstats_kernel<<<48, 256, 0, stream>>>(ybuf, parts);
    bn_finalize_kernel<<<1, 64, 0, stream>>>(parts, stats);
    bn_logits_kernel<<<N_PTS / 256, 256, 0, stream>>>(
        ybuf, stats, bn_g + it * 64, bn_b + it * 64, bw_w2 + it * 192,
        bw_b2 + it * 3, wprob);
    const float* xin = (it == 0) ? x : xtmp;
    float* xout = (it == 0) ? xtmp : (float*)d_out;
    meanshift_kernel<<<N_PTS / 256, 256, 0, stream>>>(xin, wprob, xout);
  }
}
